// KANLinear_53068615910216
// MI455X (gfx1250) — matrix operand order repaired and checked by reading, not compiled, not run on hardware
//
#include <hip/hip_runtime.h>

typedef _Float16 f16;
typedef __attribute__((ext_vector_type(16))) _Float16 v16h;
typedef __attribute__((ext_vector_type(2)))  _Float16 h2;
typedef __attribute__((ext_vector_type(8)))  float    v8f;

#define D_DIM  256
#define O_DIM  256
#define KTOT   4352           // D*16 (interp) + D (skip)
#define NKS    (KTOT / 32)    // 136 K-steps of 32
#define BM     128
#define LSTR   260            // LDS row stride (floats); 260%64=4 -> conflict-free column reads

// ---------------------------------------------------------------------------
// Pack values[O][D][16] (f32) and skip_w[O][D] (f32) into f16 WMMA B-fragment
// order.  B operand layout for v_wmma_f32_16x16x32_f16 (32x16, 16-bit):
//   lane L: N = L%16, element i holds K_local = (L/16)*8 + i (i<8), 16 + (L/16)*8 + (i-8) (i>=8).
// ws layout: frag[ks][nt][lane][i], 16 halfs (32B) contiguous per lane.
// ---------------------------------------------------------------------------
__global__ __launch_bounds__(256) void kan_prep(const float* __restrict__ values,
                                                const float* __restrict__ skip_w,
                                                f16* __restrict__ wsh) {
  int tid  = blockIdx.x * 256 + threadIdx.x;   // exactly NKS*16*32 threads
  int lane = tid & 31;
  int nt   = (tid >> 5) & 15;
  int ks   = tid >> 9;
  int o      = nt * 16 + (lane & 15);
  int kkbase = ks * 32 + ((lane >> 4) << 3);
  const float* src = (kkbase < 4096) ? (values + o * 4096 + kkbase)
                                     : (skip_w + o * 256 + (kkbase - 4096));
  v16h v;
#pragma unroll
  for (int i = 0; i < 16; ++i) v[i] = (f16)src[i < 8 ? i : i + 8];
  *(v16h*)(wsh + (size_t)tid * 16) = v;
}

__device__ __forceinline__ h2 habs2(h2 v) {
  unsigned u = __builtin_bit_cast(unsigned, v);
  u &= 0x7FFF7FFFu;
  return __builtin_bit_cast(h2, u);
}

__device__ __forceinline__ float tscale(float xv) {
  // t = (clamp(x,-1,1)+1)*7.5  in [0,15]
  return __builtin_fmaf(fminf(1.f, fmaxf(-1.f, xv)), 7.5f, 7.5f);
}

// Load the 8 B fragments (one K-step x 128 output cols) for this wave.
__device__ __forceinline__ void load_bfrags(v16h b[8], const f16* __restrict__ wsh,
                                            int ks, int Nb, int lane) {
  const f16* bp = wsh + ((size_t)(ks * 16 + Nb * 8) * 32 + lane) * 16;
#pragma unroll
  for (int nt = 0; nt < 8; ++nt) b[nt] = *(const v16h*)(bp + nt * 512);
}

// Build the A fragment for one K-step from pre-scaled t values in LDS.
//   ks < 128 : coeff(t,k) = max(0, 1 - |t - k|)  (hat basis, packed f16)
//   ks >= 128: A element  = xc = t/7.5 - 1       (skip connection)
__device__ __forceinline__ v16h build_a(const float* __restrict__ tls,
                                        int rb, int koff, int ks) {
  v16h a;
  if (ks < 128) {
    float2 tp = *(const float2*)&tls[rb * LSTR + 2 * ks];
    float d[16];
#pragma unroll
    for (int j = 0; j < 8; ++j) {
      d[j]     = tp.x - (float)(koff + j);
      d[8 + j] = tp.y - (float)(koff + j);
    }
    const h2 hone  = { (f16)1, (f16)1 };
    const h2 hzero = { (f16)0, (f16)0 };
#pragma unroll
    for (int m = 0; m < 8; ++m) {
      h2 p  = __builtin_bit_cast(h2, __builtin_amdgcn_cvt_pkrtz(d[2 * m], d[2 * m + 1]));
      h2 hv = hone - habs2(p);                       // 1 - |t-k| (packed f16)
      hv    = __builtin_elementwise_max(hv, hzero);  // v_pk_max_num_f16
      a[2 * m] = hv.x; a[2 * m + 1] = hv.y;
    }
  } else {
    const float* p0 = &tls[rb * LSTR + (ks - 128) * 32 + koff];
    const float inv = 1.0f / 7.5f;
#pragma unroll
    for (int m = 0; m < 4; ++m) {
      float2 v0 = *(const float2*)(p0 + 2 * m);
      float2 v1 = *(const float2*)(p0 + 16 + 2 * m);
      h2 pa = __builtin_bit_cast(h2, __builtin_amdgcn_cvt_pkrtz(
                  __builtin_fmaf(v0.x, inv, -1.f), __builtin_fmaf(v0.y, inv, -1.f)));
      h2 pb = __builtin_bit_cast(h2, __builtin_amdgcn_cvt_pkrtz(
                  __builtin_fmaf(v1.x, inv, -1.f), __builtin_fmaf(v1.y, inv, -1.f)));
      a[2 * m]     = pa.x; a[2 * m + 1] = pa.y;
      a[8 + 2 * m] = pb.x; a[9 + 2 * m] = pb.y;
    }
  }
  return a;
}

// ---------------------------------------------------------------------------
// GEMM: M=8192, N=256, K=4352.  Block = 128x128, 8 waves; wave w owns rows
// [Mb+16w, +16) x 128 cols (8 N-tiles).  B fragments are register
// double-buffered with the ks-loop unrolled by 2 so loads for step ks+1/ks+2
// are in flight while the 8-WMMA chain for step ks executes.
// ---------------------------------------------------------------------------
__global__ __launch_bounds__(256) void kan_gemm(const float* __restrict__ x,
                                                const f16* __restrict__ wsh,
                                                const float* __restrict__ skip_b,
                                                float* __restrict__ out) {
  extern __shared__ float tls[];          // [BM][LSTR]
  const int tid = threadIdx.x;
  const int Mb  = blockIdx.x * BM;
  const int Nb  = blockIdx.y;             // 0..1 -> 128 output cols each

  // Stage pre-scaled t for this block's 128 rows (coalesced float4 in/out).
  for (int idx = tid; idx < BM * (D_DIM / 4); idx += 256) {
    int row  = idx >> 6;
    int col4 = (idx & 63) << 2;
    float4 v = *(const float4*)(x + (size_t)(Mb + row) * D_DIM + col4);
    float4 t;
    t.x = tscale(v.x); t.y = tscale(v.y); t.z = tscale(v.z); t.w = tscale(v.w);
    *(float4*)&tls[row * LSTR + col4] = t;   // (260*row+col4)%4==0 -> aligned
  }
  __syncthreads();

  const int wid  = tid >> 5;
  const int lane = tid & 31;
  const int rb   = wid * 16 + (lane & 15);  // LDS row feeding A (A: M = lane%16)
  const int koff = (lane >> 4) << 3;        // A layout: upper lanes hold K%16 in [8,16)

  v8f  acc[8] = {};
  v16h b0[8], b1[8];

  load_bfrags(b0, wsh, 0, Nb, lane);

  for (int ks = 0; ks < NKS; ks += 2) {
    // -- step ks: prefetch ks+1, compute on b0 --
    load_bfrags(b1, wsh, ks + 1, Nb, lane);
    v16h a0 = build_a(tls, rb, koff, ks);
#pragma unroll
    for (int nt = 0; nt < 8; ++nt)
      acc[nt] = __builtin_amdgcn_wmma_f32_16x16x32_f16(false, a0, false, b0[nt],
                                                       (short)0, acc[nt], false, false);
    // -- step ks+1: prefetch ks+2 (wraps harmlessly on last iter), compute on b1 --
    int ksn = (ks + 2 < NKS) ? (ks + 2) : 0;
    load_bfrags(b0, wsh, ksn, Nb, lane);
    v16h a1 = build_a(tls, rb, koff, ks + 1);
#pragma unroll
    for (int nt = 0; nt < 8; ++nt)
      acc[nt] = __builtin_amdgcn_wmma_f32_16x16x32_f16(false, a1, false, b1[nt],
                                                       (short)0, acc[nt], false, false);
  }

  // C layout: VGPR r -> row = r + 8*(lane/16), col = lane%16.  Add bias, store f32.
  const int r0 = Mb + wid * 16 + ((lane >> 4) << 3);
#pragma unroll
  for (int nt = 0; nt < 8; ++nt) {
    int col    = Nb * 128 + nt * 16 + (lane & 15);
    float bias = skip_b[col];
#pragma unroll
    for (int r = 0; r < 8; ++r) {
      out[(size_t)(r0 + r) * O_DIM + col] = acc[nt][r] + bias;
    }
  }
}

extern "C" void kernel_launch(void* const* d_in, const int* in_sizes, int n_in,
                              void* d_out, int out_size, void* d_ws, size_t ws_size,
                              hipStream_t stream) {
  const float* x      = (const float*)d_in[0];
  const float* values = (const float*)d_in[1];
  const float* skip_w = (const float*)d_in[2];
  const float* skip_b = (const float*)d_in[3];
  // d_in[4] (grid) unused: it is linspace(-1,1,16); coefficients are closed-form.
  f16* wsh = (f16*)d_ws;   // needs KTOT*256*2 = 2,228,224 bytes

  kan_prep<<<(NKS * 16 * 32) / 256, 256, 0, stream>>>(values, skip_w, wsh);

  dim3 grid(8192 / BM, 2);
  kan_gemm<<<grid, 256, BM * LSTR * sizeof(float), stream>>>(x, wsh, skip_b, (float*)d_out);
}
